// BaseMemory_816043786284
// MI455X (gfx1250) — matrix operand order repaired and checked by reading, not compiled, not run on hardware
//
#include <hip/hip_runtime.h>
#include <cstdint>

typedef __attribute__((ext_vector_type(16))) __bf16 v16bf;
typedef __bf16 v16bf_u __attribute__((ext_vector_type(16), aligned(16)));
typedef __attribute__((ext_vector_type(8)))  float  v8f;
typedef __attribute__((ext_vector_type(4)))  float  v4f;
typedef __attribute__((ext_vector_type(4)))  unsigned int u32x4;
typedef __attribute__((ext_vector_type(8)))  int    i32x8;
typedef __attribute__((ext_vector_type(4)))  int    i32x4;

#define M_ROWS 65536
#define DDIM   768
#define EDIM   20
#define HDIM   64
#define KSTEPS 24            // 768 / 32

// B storage: per (t,g) block = 32 lanes * 48 B (32 B data + 16 B pad for
// conflict-free ds_load_b128: bank start = 12*lane mod 64 covers each 4-bank
// group exactly twice -> minimum 2-phase LDS access).
#define LANE_STRIDE_B 48
#define BLK_BYTES     (32 * LANE_STRIDE_B)          // 1536 B per (t,g)
#define BMAT_BYTES    (KSTEPS * 4 * BLK_BYTES)      // 147456 B per hi/lo
#define WS_BHI   0
#define WS_BLO   BMAT_BYTES                         // 147456
#define WS_BBYTES (2 * BMAT_BYTES)                  // 294912 (one TDM copy)
#define WS_C0    294912                             // 64 f32
#define WS_DISTH 295168                             // 10*64 f32
#define WS_CNTH  297728                             // 10*64 f32

__device__ __forceinline__ int bucketf(int c) {
  int cc = c < 1 ? 1 : c;
  int lg = 31 - __clz(cc);                 // floor(log2(max(c,1)))
  int idx = (c <= 4) ? c : (lg + 3);
  idx = idx < 0 ? 0 : idx;
  return idx > 9 ? 9 : idx;
}

// ---------------------------------------------------------------------------
// Prep: fold ment into W1 (Wc = W1a + ment*W1c), split into bf16 hi/lo packed
// in the WMMA B-operand lane layout (48 B/lane padded rows); precompute
// c0, DistH, CntH.
// ---------------------------------------------------------------------------
__global__ void bm_prep_kernel(const float* __restrict__ ment,
                               const float* __restrict__ W1,
                               const float* __restrict__ b1,
                               const float* __restrict__ dist_table,
                               const float* __restrict__ counter_table,
                               __bf16* __restrict__ Bhi, __bf16* __restrict__ Blo,
                               float* __restrict__ c0,
                               float* __restrict__ DistH, float* __restrict__ CntH) {
  const int tid = threadIdx.x;

  // logical idx = (((t*4 + g)*32 + lane)*16 + e)
  // B layout (16-bit B 32x16): lane<16 -> N=lane,    K = e<8 ? e : 8+e
  //                            lane>=16 -> N=lane-16, K = e<8 ? 8+e : 16+e
  for (int idx = tid; idx < KSTEPS * 4 * 32 * 16; idx += blockDim.x) {
    int e    = idx & 15;
    int lane = (idx >> 4) & 31;
    int g    = (idx >> 9) & 3;
    int t    = idx >> 11;
    int n    = lane & 15;
    int klocal = ((lane < 16) ? 0 : 8) + ((e < 8) ? e : 8 + e);
    int k = t * 32 + klocal;
    int h = g * 16 + n;
    float wc = W1[k * HDIM + h] + ment[k] * W1[(2 * DDIM + k) * HDIM + h];
    __bf16 hi = (__bf16)wc;
    float  lo = wc - (float)hi;
    // padded storage position (in bf16 elements)
    int sidx = (t * 4 + g) * (BLK_BYTES / 2) + lane * (LANE_STRIDE_B / 2) + e;
    Bhi[sidx] = hi;
    Blo[sidx] = (__bf16)lo;
  }

  for (int h = tid; h < HDIM; h += blockDim.x) {
    float s = b1[h];
    for (int d = 0; d < DDIM; ++d) s += ment[d] * W1[(DDIM + d) * HDIM + h];
    c0[h] = s;
  }

  for (int idx = tid; idx < 10 * HDIM; idx += blockDim.x) {
    int b = idx / HDIM, h = idx - b * HDIM;
    float s1 = 0.f, s2 = 0.f;
    for (int e = 0; e < EDIM; ++e) {
      s1 += dist_table[b * EDIM + e]    * W1[(3 * DDIM + e) * HDIM + h];
      s2 += counter_table[b * EDIM + e] * W1[(3 * DDIM + EDIM + e) * HDIM + h];
    }
    DistH[idx] = s1;
    CntH[idx]  = s2;
  }
}

// float32 -> (bf16 hi, bf16 lo) split for one A operand (16 values / lane)
__device__ __forceinline__ void cvt_split(const v4f x0, const v4f x1,
                                          const v4f x2, const v4f x3,
                                          v16bf& hi, v16bf& lo) {
#pragma unroll
  for (int e = 0; e < 16; ++e) {
    float v = (e < 4) ? x0[e] : (e < 8) ? x1[e - 4] : (e < 12) ? x2[e - 8] : x3[e - 12];
    __bf16 h = (__bf16)v;
    hi[e] = h;
    lo[e] = (__bf16)(v - (float)h);
  }
}

// epilogue for one 16-row tile held in acc[0..3]
__device__ __forceinline__ void epilogue_tile(const v8f* acc, int rowbase, int lane,
    const float* __restrict__ c0, const float* __restrict__ DistH,
    const float* __restrict__ CntH, const float* __restrict__ W2, float b2v,
    const int* __restrict__ ent_counter, const int* __restrict__ last_mention,
    int ms, float* __restrict__ out) {
  const int n   = lane & 15;
  const int hih = lane >> 4;
  const int myrow = rowbase + n;
  const int lm = last_mention[myrow];
  const int ec = ent_counter[myrow];
  const int bd = bucketf(ms - lm);
  const int bc = bucketf(ec);
  float c0v[4], w2v[4];
#pragma unroll
  for (int g = 0; g < 4; ++g) { c0v[g] = c0[g * 16 + n]; w2v[g] = W2[g * 16 + n]; }
#pragma unroll
  for (int j = 0; j < 8; ++j) {
    const int rr  = hih * 8 + j;
    const int bdr = __shfl(bd, rr);
    const int bcr = __shfl(bc, rr);
    float s = 0.f;
#pragma unroll
    for (int g = 0; g < 4; ++g) {
      const int col = g * 16 + n;
      float hv = acc[g][j] + c0v[g] + DistH[bdr * HDIM + col] + CntH[bcr * HDIM + col];
      s += fmaxf(hv, 0.f) * w2v[g];
    }
    s += __shfl_xor(s, 1);
    s += __shfl_xor(s, 2);
    s += __shfl_xor(s, 4);
    s += __shfl_xor(s, 8);
    const int ecr = __shfl(ec, rr);
    if (n == j) out[rowbase + rr] = (ecr > 0) ? (s + b2v) : -10000.0f;
  }
}

// ---------------------------------------------------------------------------
// Main: one wave per 32 rows (two 16-row tiles sharing B reads).
// Folded weights staged into LDS once per block via the Tensor Data Mover.
// ---------------------------------------------------------------------------
__global__ __launch_bounds__(256)
void bm_score_kernel(const float* __restrict__ mem,
                     const __bf16* __restrict__ Bglob,   // Bhi|Blo, 294912 B
                     const float* __restrict__ c0,
                     const float* __restrict__ DistH, const float* __restrict__ CntH,
                     const float* __restrict__ W2, const float* __restrict__ b2,
                     const int* __restrict__ ent_counter,
                     const int* __restrict__ last_mention,
                     const int* __restrict__ ment_start,
                     float* __restrict__ out) {
  extern __shared__ char smem[];                    // 294912 B dynamic LDS

  // ---- stage Bhi|Blo into LDS (TDM, wave 0 only; TDM ignores EXEC) ----
#if __has_builtin(__builtin_amdgcn_tensor_load_to_lds) && __has_builtin(__builtin_amdgcn_s_wait_tensorcnt)
  if (threadIdx.x < 32) {
    uint64_t ga  = (uint64_t)(uintptr_t)Bglob;
    uint32_t lds = (uint32_t)(uintptr_t)smem;
    // D# group0: count=1 | lds_addr | global_addr[56:0] | type=2
    u32x4 g0 = { 1u, lds, (uint32_t)ga,
                 (uint32_t)((ga >> 32) & 0x01FFFFFFu) | 0x80000000u };
    // D# group1: data_size=8B; 1D copy: tensor_dim0=tile_dim0=stride0=36864
    i32x8 g1 = { 0x30000,             // workgroup_mask=0, data_size=3 (8B)
                 (int)0x90000000,     // tensor_dim0[15:0]=36864 << 16
                 0x10000,             // tensor_dim0 hi=0 | tensor_dim1=1
                 (int)0x90000000,     // tensor_dim1 hi=0 | tile_dim0=36864
                 1,                   // tile_dim1=1, tile_dim2=0
                 36864, 0, 0 };       // tensor_dim0_stride=36864
    i32x4 gz4 = { 0, 0, 0, 0 };       // groups 2/3 unused (<=2D tensor)
    i32x8 gz8 = { 0, 0, 0, 0, 0, 0, 0, 0 };
    __builtin_amdgcn_tensor_load_to_lds(g0, g1, gz4, gz4, gz8, 0);
    __builtin_amdgcn_s_wait_tensorcnt((short)0);
  }
#else
  for (int i = threadIdx.x; i < WS_BBYTES / 16; i += blockDim.x)
    ((int4*)smem)[i] = ((const int4*)Bglob)[i];
#endif
  __syncthreads();

  const int lane = threadIdx.x & 31;
  const int wave = threadIdx.x >> 5;
  const int row0 = (blockIdx.x * 8 + wave) * 32;    // 32 rows per wave
  const int n    = lane & 15;
  const int hih  = lane >> 4;
  const int kb   = hih * 8;                         // A-operand K base

  const char* sB_lane = smem + lane * LANE_STRIDE_B;  // per-lane padded base

  const float* arow0 = mem + (size_t)(row0 + n) * DDIM;
  const float* arow1 = mem + (size_t)(row0 + 16 + n) * DDIM;

  v8f acc0[4], acc1[4];
#pragma unroll
  for (int g = 0; g < 4; ++g) { acc0[g] = (v8f){}; acc1[g] = (v8f){}; }

  // software-pipelined A: preload K-step 0
  v4f a00 = *(const v4f*)(arow0 + kb);
  v4f a01 = *(const v4f*)(arow0 + kb + 4);
  v4f a02 = *(const v4f*)(arow0 + kb + 16);
  v4f a03 = *(const v4f*)(arow0 + kb + 20);
  v4f a10 = *(const v4f*)(arow1 + kb);
  v4f a11 = *(const v4f*)(arow1 + kb + 4);
  v4f a12 = *(const v4f*)(arow1 + kb + 16);
  v4f a13 = *(const v4f*)(arow1 + kb + 20);

  for (int t = 0; t < KSTEPS; ++t) {
    v16bf ahi0, alo0, ahi1, alo1;
    cvt_split(a00, a01, a02, a03, ahi0, alo0);
    cvt_split(a10, a11, a12, a13, ahi1, alo1);

    if (t + 1 < KSTEPS) {                 // register-preload next K-step
      const int K1 = (t + 1) * 32;
      a00 = *(const v4f*)(arow0 + K1 + kb);
      a01 = *(const v4f*)(arow0 + K1 + kb + 4);
      a02 = *(const v4f*)(arow0 + K1 + kb + 16);
      a03 = *(const v4f*)(arow0 + K1 + kb + 20);
      a10 = *(const v4f*)(arow1 + K1 + kb);
      a11 = *(const v4f*)(arow1 + K1 + kb + 4);
      a12 = *(const v4f*)(arow1 + K1 + kb + 16);
      a13 = *(const v4f*)(arow1 + K1 + kb + 20);
      __builtin_prefetch(arow0 + K1 + 32, 0, 3);   // L2-warm K-step t+2
      __builtin_prefetch(arow1 + K1 + 32, 0, 3);
    }

    const char* bt = sB_lane + (size_t)t * (4 * BLK_BYTES);
#pragma unroll
    for (int g = 0; g < 4; ++g) {
      v16bf bh = *(const v16bf_u*)(bt + g * BLK_BYTES);
      v16bf bl = *(const v16bf_u*)(bt + g * BLK_BYTES + BMAT_BYTES);
      acc0[g] = __builtin_amdgcn_wmma_f32_16x16x32_bf16(false, ahi0, false, bh,
                                                        (short)0, acc0[g], false, false);
      acc1[g] = __builtin_amdgcn_wmma_f32_16x16x32_bf16(false, ahi1, false, bh,
                                                        (short)0, acc1[g], false, false);
      acc0[g] = __builtin_amdgcn_wmma_f32_16x16x32_bf16(false, alo0, false, bh,
                                                        (short)0, acc0[g], false, false);
      acc1[g] = __builtin_amdgcn_wmma_f32_16x16x32_bf16(false, alo1, false, bh,
                                                        (short)0, acc1[g], false, false);
      acc0[g] = __builtin_amdgcn_wmma_f32_16x16x32_bf16(false, ahi0, false, bl,
                                                        (short)0, acc0[g], false, false);
      acc1[g] = __builtin_amdgcn_wmma_f32_16x16x32_bf16(false, ahi1, false, bl,
                                                        (short)0, acc1[g], false, false);
    }
  }

  // ---- epilogue ----
  const int ms  = ment_start[0];
  const float b2v = b2[0];
  epilogue_tile(acc0, row0,      lane, c0, DistH, CntH, W2, b2v,
                ent_counter, last_mention, ms, out);
  epilogue_tile(acc1, row0 + 16, lane, c0, DistH, CntH, W2, b2v,
                ent_counter, last_mention, ms, out);

  if (blockIdx.x == 0 && threadIdx.x == 0) out[M_ROWS] = 0.0f;  // dummy slot
}

// ---------------------------------------------------------------------------
extern "C" void kernel_launch(void* const* d_in, const int* in_sizes, int n_in,
                              void* d_out, int out_size, void* d_ws, size_t ws_size,
                              hipStream_t stream) {
  const float* ment          = (const float*)d_in[0];
  const float* mem           = (const float*)d_in[1];
  const float* dist_table    = (const float*)d_in[2];
  const float* counter_table = (const float*)d_in[3];
  const float* W1            = (const float*)d_in[4];
  const float* b1            = (const float*)d_in[5];
  const float* W2            = (const float*)d_in[6];
  const float* b2            = (const float*)d_in[7];
  const int*   ent_counter   = (const int*)d_in[8];
  const int*   last_mention  = (const int*)d_in[9];
  const int*   ment_start    = (const int*)d_in[10];
  float* out = (float*)d_out;

  char* ws = (char*)d_ws;
  __bf16* Bhi  = (__bf16*)(ws + WS_BHI);
  __bf16* Blo  = (__bf16*)(ws + WS_BLO);
  float* c0    = (float*)(ws + WS_C0);
  float* DistH = (float*)(ws + WS_DISTH);
  float* CntH  = (float*)(ws + WS_CNTH);

  bm_prep_kernel<<<1, 256, 0, stream>>>(ment, W1, b1, dist_table, counter_table,
                                        Bhi, Blo, c0, DistH, CntH);

  // 65536 rows / (8 waves * 32 rows) = 256 blocks; 288 KB dynamic LDS
  bm_score_kernel<<<256, 256, WS_BBYTES, stream>>>(mem, Bhi, c0, DistH, CntH,
                                                   W2, b2, ent_counter, last_mention,
                                                   ment_start, out);
}
